// GCN_Dense_Att_6176162972210
// MI455X (gfx1250) — compile-verified, matrix operand order checked by reading, not run on hardware
//
#include <hip/hip_runtime.h>
#include <hip/hip_bf16.h>

typedef __attribute__((ext_vector_type(16))) __bf16 v16bf;
typedef __attribute__((ext_vector_type(8)))  float  v8f;

#define GCN_IN   300
#define GCN_INP  320   // padded K for layer 1 (mult of 32)
#define GCN_HID  256
#define GCN_D    3
#define NEG_SLOPE 0.2f
#define SPMM_EPW 4     // edges per wave

union FragBF { v16bf v; unsigned int u[8]; };

// split f into bf16 hi + bf16 lo(residual); pack two k-consecutive values per
// dword with the even-k value in bits [15:0] (matches A/B VGPR pair layout).
__device__ __forceinline__ void split_pack(float f0, float f1,
                                           unsigned int& hi, unsigned int& lo) {
  const __bf16 h0 = (__bf16)f0, h1 = (__bf16)f1;
  const __bf16 l0 = (__bf16)(f0 - (float)h0), l1 = (__bf16)(f1 - (float)h1);
  hi = ((unsigned int)__builtin_bit_cast(unsigned short, h1) << 16) |
       (unsigned int)__builtin_bit_cast(unsigned short, h0);
  lo = ((unsigned int)__builtin_bit_cast(unsigned short, l1) << 16) |
       (unsigned int)__builtin_bit_cast(unsigned short, l0);
}

// pair index d (0..7) -> kbase within a 32-wide K step for a given khalf
__device__ __forceinline__ int kbase_of(int d, int khalf) {
  return ((d < 4) ? (2 * d) : (16 + 2 * (d - 4))) + 8 * khalf;
}

// ---------------------------------------------------------------------------
// softmax over the two 3-element attention vectors -> att[0..2], att[3..5]
// ---------------------------------------------------------------------------
__global__ void softmax3_kernel(const float* __restrict__ a_att,
                                const float* __restrict__ r_att,
                                float* __restrict__ att) {
  if (threadIdx.x == 0 && blockIdx.x == 0) {
    for (int s = 0; s < 2; ++s) {
      const float* v = (s == 0) ? a_att : r_att;
      float m = v[0];
      for (int i = 1; i < GCN_D; ++i) m = fmaxf(m, v[i]);
      float e[GCN_D], sum = 0.f;
      for (int i = 0; i < GCN_D; ++i) { e[i] = __expf(v[i] - m); sum += e[i]; }
      for (int i = 0; i < GCN_D; ++i) att[s * GCN_D + i] = e[i] / sum;
    }
  }
}

// ---------------------------------------------------------------------------
// pre-pass: W[K x 256] fp32 -> fragment-major bf16 hi/lo tiles, zero-padded
// to KP rows.  Layout: [kstep][coltile(16)][part(2)][lane(32)] x uint4, so a
// wave's B half-fragment is 32 consecutive uint4s (one coalesced b128/lane).
// One uint4 (hi) + one uint4 (lo) per thread; grid-stride over KP*32 uint4s.
// ---------------------------------------------------------------------------
__global__ void pack_w_kernel(const float* __restrict__ w,
                              uint4* __restrict__ whi,
                              uint4* __restrict__ wlo,
                              int K, int KP) {
  const int total = KP * 32;               // uint4 count
  int g = blockIdx.x * blockDim.x + threadIdx.x;
  const int stride = gridDim.x * blockDim.x;
  for (; g < total; g += stride) {
    const int lane = g & 31;
    int rest = g >> 5;
    const int part = rest & 1;  rest >>= 1;
    const int tile = rest & 15; rest >>= 4;
    const int ks   = rest;                  // K-step
    const int l15   = lane & 15;
    const int khalf = lane >> 4;
    const int col   = tile * 16 + l15;
    unsigned int hi[4], lo[4];
    #pragma unroll
    for (int q = 0; q < 4; ++q) {
      const int d = part * 4 + q;
      const int k = ks * 32 + kbase_of(d, khalf);
      const float f0 = (k     < K) ? w[(size_t)k * GCN_HID + col]       : 0.f;
      const float f1 = (k + 1 < K) ? w[(size_t)(k + 1) * GCN_HID + col] : 0.f;
      split_pack(f0, f1, hi[q], lo[q]);
    }
    whi[g] = make_uint4(hi[0], hi[1], hi[2], hi[3]);
    wlo[g] = make_uint4(lo[0], lo[1], lo[2], lo[3]);
  }
}

// ---------------------------------------------------------------------------
// grid-stride float4 zero fill (n multiple of 4)
// ---------------------------------------------------------------------------
__global__ void zero_kernel(float4* __restrict__ p, size_t n4) {
  size_t i = (size_t)blockIdx.x * blockDim.x + threadIdx.x;
  size_t stride = (size_t)gridDim.x * blockDim.x;
  const float4 z = make_float4(0.f, 0.f, 0.f, 0.f);
  for (; i < n4; i += stride) p[i] = z;
}

// ---------------------------------------------------------------------------
// WMMA GEMM:  C[M x 256] = A[M x KR] @ W[KR x 256] + bias
//   KP: padded K (mult of 32) = extent of packed W;  KR: real K (A's ld)
//   Block = 128 threads (4 waves) -> 16x256 slab.
//   A staged to LDS pre-split/pair-packed; W pre-split fragment-major (bf16).
//   Inner loop: 4 ds_load_b64x2 + 16 global b128 + 12 wmma per K-step.
// ---------------------------------------------------------------------------
template <int KP, int KR>
__global__ void gemm_bias_wmma(const float* __restrict__ A,
                               const uint4* __restrict__ Whi,
                               const uint4* __restrict__ Wlo,
                               const float* __restrict__ bias,
                               float* __restrict__ C,
                               int M) {
  constexpr int NCOL = GCN_HID;     // 256
  constexpr int LDSW = 18;          // 16 dwords + 2 pad: conflict-free
  __shared__ unsigned int AsHi[16 * LDSW];
  __shared__ unsigned int AsLo[16 * LDSW];

  const int tid   = threadIdx.x;
  const int lane  = tid & 31;
  const int wave  = tid >> 5;
  const int mtile = blockIdx.x;
  const int l15   = lane & 15;
  const int khalf = lane >> 4;      // 0: K 0-7,16-23   1: K 8-15,24-31

  // A staging: thread t handles A[row = t/8, k = (t%8)*4 .. +3]
  const int arow  = tid >> 3;       // 0..15
  const int acol  = (tid & 7) * 4;  // 0,4,...,28
  const int gr    = mtile * 16 + arow;
  const int garow = (gr < M) ? gr : (M - 1);

  v8f acc[4];
  #pragma unroll
  for (int j = 0; j < 4; ++j)
    #pragma unroll
    for (int i = 0; i < 8; ++i) acc[j][i] = 0.f;

  #pragma unroll 1
  for (int k0 = 0; k0 < KP; k0 += 32) {
    // ---- load + split A quad (only the KR%32 tail carries guards) ----
    float a4[4];
    if (KP == KR || k0 + 32 <= KR) {
      const float4 t4 =
          *reinterpret_cast<const float4*>(A + (size_t)garow * KR + k0 + acol);
      a4[0] = t4.x; a4[1] = t4.y; a4[2] = t4.z; a4[3] = t4.w;
    } else {
      #pragma unroll
      for (int i = 0; i < 4; ++i) {
        const int k = k0 + acol + i;
        a4[i] = (k < KR) ? A[(size_t)garow * KR + k] : 0.f;
      }
    }
    unsigned int hi0, lo0, hi1, lo1;
    split_pack(a4[0], a4[1], hi0, lo0);
    split_pack(a4[2], a4[3], hi1, lo1);

    __syncthreads();                       // previous step's reads done
    AsHi[arow * LDSW + acol / 2]     = hi0;
    AsHi[arow * LDSW + acol / 2 + 1] = hi1;
    AsLo[arow * LDSW + acol / 2]     = lo0;
    AsLo[arow * LDSW + acol / 2 + 1] = lo1;
    __syncthreads();                       // tile visible to all waves

    // ---- A fragments: dword reads from LDS (-> ds_load_2addr_b64) ----
    FragBF a_hi, a_lo;
    #pragma unroll
    for (int v = 0; v < 8; ++v) {
      const int dw = ((v < 4) ? v : v + 4) + 4 * khalf;  // pair index
      a_hi.u[v] = AsHi[l15 * LDSW + dw];
      a_lo.u[v] = AsLo[l15 * LDSW + dw];
    }

    // ---- 4 column tiles per wave; B fragments: 2x b128 each (coalesced) ----
    const size_t stepBase = (size_t)(k0 >> 5) * (16 * 2 * 32);
    #pragma unroll
    for (int j = 0; j < 4; ++j) {
      const int tile = wave * 4 + j;
      const size_t base = stepBase + (size_t)tile * (2 * 32) + lane;
      FragBF b_hi, b_lo;
      const uint4 h0 = Whi[base], h1 = Whi[base + 32];
      const uint4 g0 = Wlo[base], g1 = Wlo[base + 32];
      b_hi.u[0] = h0.x; b_hi.u[1] = h0.y; b_hi.u[2] = h0.z; b_hi.u[3] = h0.w;
      b_hi.u[4] = h1.x; b_hi.u[5] = h1.y; b_hi.u[6] = h1.z; b_hi.u[7] = h1.w;
      b_lo.u[0] = g0.x; b_lo.u[1] = g0.y; b_lo.u[2] = g0.z; b_lo.u[3] = g0.w;
      b_lo.u[4] = g1.x; b_lo.u[5] = g1.y; b_lo.u[6] = g1.z; b_lo.u[7] = g1.w;
      // hi*hi + hi*lo + lo*hi (lo*lo ~2^-16, dropped)
      acc[j] = __builtin_amdgcn_wmma_f32_16x16x32_bf16(
          false, a_hi.v, false, b_hi.v, (short)0, acc[j], false, false);
      acc[j] = __builtin_amdgcn_wmma_f32_16x16x32_bf16(
          false, a_hi.v, false, b_lo.v, (short)0, acc[j], false, false);
      acc[j] = __builtin_amdgcn_wmma_f32_16x16x32_bf16(
          false, a_lo.v, false, b_hi.v, (short)0, acc[j], false, false);
    }
  }

  // ---- store D tiles (+bias). C/D layout: VGPR v -> M = v + 8*khalf ----
  #pragma unroll
  for (int j = 0; j < 4; ++j) {
    const int col = wave * 64 + j * 16 + l15;
    const float bv = bias[col];
    #pragma unroll
    for (int v = 0; v < 8; ++v) {
      const int r = mtile * 16 + v + khalf * 8;
      if (r < M) C[(size_t)r * NCOL + col] = acc[j][v] + bv;
    }
  }
}

// ---------------------------------------------------------------------------
// SPMM scatter:  out[rows[e]] += att[rel] * vals[e] * in[cols[e]]
// One wave per SPMM_EPW consecutive edges; float4 gathers (b128), prefetch of
// the next edge's row. Feature rows (51MB) are L2-resident in the 192MB L2,
// so this is L2-latency/atomic-throughput bound.
// ---------------------------------------------------------------------------
__global__ void spmm_scatter(const int*   __restrict__ rows,
                             const int*   __restrict__ cols,
                             const float* __restrict__ vals,
                             const float* __restrict__ att,
                             const float* __restrict__ in,
                             float*       __restrict__ out,
                             int E, long long totalEdges) {
  const long long wid =
      ((long long)blockIdx.x * blockDim.x + threadIdx.x) >> 5;
  const int lane = threadIdx.x & 31;
  const long long base = wid * SPMM_EPW;
  if (base >= totalEdges) return;

  #pragma unroll
  for (int t = 0; t < SPMM_EPW; ++t) {
    const long long e = base + t;
    if (e >= totalEdges) break;
    // prefetch next edge's source row (covers all 8 cachelines of the 1KB row)
    if (t + 1 < SPMM_EPW && e + 1 < totalEdges) {
      const int nc = cols[e + 1];
      __builtin_prefetch(in + (size_t)nc * GCN_HID + lane * 8, 0, 0);
    }
    const int rel = (int)(e / E);
    const float v = att[rel] * vals[e];
    const float4* __restrict__ s4 =
        reinterpret_cast<const float4*>(in + (size_t)cols[e] * GCN_HID);
    float* __restrict__ drow = out + (size_t)rows[e] * GCN_HID;
    #pragma unroll
    for (int i = 0; i < 2; ++i) {
      const float4 x4 = s4[lane * 2 + i];
      const int c = (lane * 2 + i) * 4;
      atomicAdd(&drow[c + 0], v * x4.x);
      atomicAdd(&drow[c + 1], v * x4.y);
      atomicAdd(&drow[c + 2], v * x4.z);
      atomicAdd(&drow[c + 3], v * x4.w);
    }
  }
}

// ---------------------------------------------------------------------------
// LeakyReLU(0.2) in place, float4
// ---------------------------------------------------------------------------
__global__ void leaky_relu_kernel(float4* __restrict__ p, size_t n4) {
  size_t i = (size_t)blockIdx.x * blockDim.x + threadIdx.x;
  size_t stride = (size_t)gridDim.x * blockDim.x;
  for (; i < n4; i += stride) {
    float4 x = p[i];
    x.x = (x.x > 0.f) ? x.x : NEG_SLOPE * x.x;
    x.y = (x.y > 0.f) ? x.y : NEG_SLOPE * x.y;
    x.z = (x.z > 0.f) ? x.z : NEG_SLOPE * x.z;
    x.w = (x.w > 0.f) ? x.w : NEG_SLOPE * x.w;
    p[i] = x;
  }
}

// ---------------------------------------------------------------------------
// Row-wise L2 normalize (eps 1e-12). One wave per row, float4 + shfl_xor.
// ---------------------------------------------------------------------------
__global__ void l2norm_kernel(float* __restrict__ p, int N) {
  const int wave = threadIdx.x >> 5;
  const int lane = threadIdx.x & 31;
  const int row  = blockIdx.x * (blockDim.x >> 5) + wave;
  if (row >= N) return;
  float4* r4 = reinterpret_cast<float4*>(p + (size_t)row * GCN_HID);
  float4 x[2];
  float ss = 0.f;
  #pragma unroll
  for (int i = 0; i < 2; ++i) {
    x[i] = r4[lane * 2 + i];
    ss += x[i].x * x[i].x + x[i].y * x[i].y + x[i].z * x[i].z + x[i].w * x[i].w;
  }
  #pragma unroll
  for (int off = 16; off > 0; off >>= 1) ss += __shfl_xor(ss, off, 32);
  const float sc = 1.f / fmaxf(sqrtf(ss), 1e-12f);
  #pragma unroll
  for (int i = 0; i < 2; ++i) {
    x[i].x *= sc; x[i].y *= sc; x[i].z *= sc; x[i].w *= sc;
    r4[lane * 2 + i] = x[i];
  }
}

// ---------------------------------------------------------------------------
extern "C" void kernel_launch(void* const* d_in, const int* in_sizes, int n_in,
                              void* d_out, int out_size, void* d_ws, size_t ws_size,
                              hipStream_t stream) {
  const float* x      = (const float*)d_in[0];
  const float* w1     = (const float*)d_in[1];
  const float* b1     = (const float*)d_in[2];
  const float* w2     = (const float*)d_in[3];
  const float* b2     = (const float*)d_in[4];
  const float* a_att  = (const float*)d_in[5];
  const float* r_att  = (const float*)d_in[6];
  const int*   src    = (const int*)d_in[7];
  const int*   dst    = (const int*)d_in[8];
  const float* vals_a = (const float*)d_in[9];
  const float* vals_r = (const float*)d_in[10];
  float* out = (float*)d_out;

  const int N = in_sizes[0] / GCN_IN;      // 50000
  const int E = in_sizes[7] / GCN_D;       // 800000
  const size_t nh = (size_t)N * GCN_HID;   // 12.8M floats

  // ---- workspace layout ----
  float* s    = (float*)d_ws;              // [N, HID]
  float* h    = s + nh;                    // [N, HID]
  float* att  = h + nh;                    // [6] (16 reserved, keeps 16B align)
  uint4* whi1 = (uint4*)(att + 16);        // [GCN_INP*32] uint4
  uint4* wlo1 = whi1 + (size_t)GCN_INP * 32;
  uint4* whi2 = wlo1 + (size_t)GCN_INP * 32;  // [GCN_HID*32] uint4
  uint4* wlo2 = whi2 + (size_t)GCN_HID * 32;

  // ---- pre-passes: attention softmax + weight split/pack ----
  softmax3_kernel<<<1, 32, 0, stream>>>(a_att, r_att, att);
  pack_w_kernel<<<40, 256, 0, stream>>>(w1, whi1, wlo1, GCN_IN,  GCN_INP);
  pack_w_kernel<<<32, 256, 0, stream>>>(w2, whi2, wlo2, GCN_HID, GCN_HID);
  zero_kernel<<<2048, 256, 0, stream>>>((float4*)h,   nh / 4);
  zero_kernel<<<2048, 256, 0, stream>>>((float4*)out, nh / 4);

  const int mblocks = (N + 15) / 16;
  const long long totalEdges = (long long)GCN_D * E;
  const long long totalWaves = (totalEdges + SPMM_EPW - 1) / SPMM_EPW;
  const int spmmBlocks = (int)((totalWaves + 7) / 8);  // 8 waves / 256-thr block

  // ---- layer 1 ----
  gemm_bias_wmma<GCN_INP, GCN_IN>
      <<<mblocks, 128, 0, stream>>>(x, whi1, wlo1, b1, s, N);
  spmm_scatter<<<spmmBlocks, 256, 0, stream>>>(dst, src, vals_a, att, s, h,
                                               E, totalEdges);
  leaky_relu_kernel<<<2048, 256, 0, stream>>>((float4*)h, nh / 4);

  // ---- layer 2 ----
  gemm_bias_wmma<GCN_HID, GCN_HID>
      <<<mblocks, 128, 0, stream>>>(h, whi2, wlo2, b2, s, N);
  spmm_scatter<<<spmmBlocks, 256, 0, stream>>>(src, dst, vals_r, att + 3, s, out,
                                               E, totalEdges);

  // ---- row-wise L2 normalize ----
  l2norm_kernel<<<(N + 7) / 8, 256, 0, stream>>>(out, N);
}
